// PrototypicalLoss_65704409694243
// MI455X (gfx1250) — matrix operand order repaired and checked by reading, not compiled, not run on hardware
//
#include <hip/hip_runtime.h>
#include <hip/hip_bf16.h>

typedef __attribute__((ext_vector_type(8)))  float  v8f;
typedef __attribute__((ext_vector_type(16))) __bf16 v16bf;

#define N_ROWS 65536
#define DIM    512
#define NCLS   64
#define BSTRIDE 520   // 512 + 8 bf16 pad: 1040 B rows -> 16B aligned, stride mod 64 dwords == 4

// float -> bf16 helpers routed through the guaranteed HIP API, bit-cast to __bf16.
__device__ __forceinline__ __bf16 f2bf(float x) {
  __hip_bfloat16 h = __float2bfloat16(x);
  return __builtin_bit_cast(__bf16, h);
}
__device__ __forceinline__ float bf2f(__bf16 b) {
  return __bfloat162float(__builtin_bit_cast(__hip_bfloat16, b));
}

// ---------------------------------------------------------------------------
// Kernel 1: deterministic per-class sums + counts (one HBM pass over E).
// grid = (64 classes, 4 column-chunks), block = 128. Sequential scan per block
// -> bitwise-deterministic fp sums; label reads are block-uniform (scalar).
// ---------------------------------------------------------------------------
__global__ void proto_sum_kernel(const float* __restrict__ E,
                                 const int* __restrict__ labels,
                                 float* __restrict__ protoSum,
                                 float* __restrict__ counts) {
  const int c   = blockIdx.x;
  const int col = blockIdx.y * 128 + threadIdx.x;
  float acc = 0.0f;
  int   cnt = 0;
  for (int r = 0; r < N_ROWS; ++r) {
    if (labels[r] == c) {
      acc += E[(size_t)r * DIM + col];
      ++cnt;
    }
  }
  protoSum[c * DIM + col] = acc;
  if (blockIdx.y == 0 && threadIdx.x == 0) counts[c] = (float)cnt;
}

// ---------------------------------------------------------------------------
// Kernel 2: prototypes = sums/counts (in place, fp32), p2[c] = ||p_c||^2,
// plus bf16 hi/lo split planes for the fast-pipe GEMM.
// ---------------------------------------------------------------------------
__global__ void proto_finalize_kernel(float* __restrict__ proto,
                                      const float* __restrict__ counts,
                                      float* __restrict__ p2,
                                      __hip_bfloat16* __restrict__ pHi,
                                      __hip_bfloat16* __restrict__ pLo) {
  const int c = blockIdx.x;
  const int t = threadIdx.x;               // 128 threads = 4 waves
  const float inv = 1.0f / counts[c];
  float s2 = 0.0f;
#pragma unroll
  for (int d = t; d < DIM; d += 128) {
    const float v = proto[c * DIM + d] * inv;
    proto[c * DIM + d] = v;
    const __hip_bfloat16 h = __float2bfloat16(v);
    pHi[c * DIM + d] = h;
    pLo[c * DIM + d] = __float2bfloat16(v - __bfloat162float(h));
    s2 += v * v;
  }
#pragma unroll
  for (int off = 1; off < 32; off <<= 1) s2 += __shfl_xor(s2, off, 32);
  __shared__ float red[4];
  if ((t & 31) == 0) red[t >> 5] = s2;
  __syncthreads();
  if (t == 0) p2[c] = red[0] + red[1] + red[2] + red[3];
}

// ---------------------------------------------------------------------------
// Kernel 3: e2[r] = ||e_r||^2 exact fp32 (one wave/row; E is L2-hot).
// ---------------------------------------------------------------------------
__global__ void e2_kernel(const float* __restrict__ E, float* __restrict__ e2) {
  const int wv   = threadIdx.x >> 5;
  const int lane = threadIdx.x & 31;
  const int row  = blockIdx.x * 8 + wv;
  const float4* rp = (const float4*)(E + (size_t)row * DIM);
  float s = 0.0f;
#pragma unroll
  for (int p = 0; p < 4; ++p) {
    float4 v = rp[lane + 32 * p];
    s += v.x * v.x + v.y * v.y + v.z * v.z + v.w * v.w;
  }
#pragma unroll
  for (int off = 1; off < 32; off <<= 1) s += __shfl_xor(s, off, 32);
  if (lane == 0) e2[row] = s;
}

// ---------------------------------------------------------------------------
// Kernel 4: split-bf16 cross-GEMM (v_wmma_f32_16x16x32_bf16) + distance +
// log-softmax + NLL/accuracy epilogue, fused. Block = 256 (8 waves), each
// wave owns a 16-row tile; bf16 hi/lo prototype planes live in padded LDS,
// staged via TDM tensor_load_to_lds when the toolchain exposes it.
// A frag (16-bit 16x32): lane half h -> contiguous K run {k+16h..+15}.
// B frag (16-bit 32x16): lane half h -> contiguous K run {k+16h..+15}.
// C/D frag (f32 16x16): vgpr j = row j (lanes 0-15) / row j+8 (lanes 16-31).
// ---------------------------------------------------------------------------
__global__ void __launch_bounds__(256)
cross_softmax_kernel(const float* __restrict__ E,
                     const int* __restrict__ labels,
                     const __hip_bfloat16* __restrict__ pHi,
                     const __hip_bfloat16* __restrict__ pLo,
                     const float* __restrict__ p2,
                     const float* __restrict__ e2,
                     float* __restrict__ partLoss,
                     float* __restrict__ partCorr) {
  __shared__ __align__(16) __bf16 BhiLds[NCLS * BSTRIDE];  // ~65 KB
  __shared__ __align__(16) __bf16 BloLds[NCLS * BSTRIDE];  // ~65 KB
  __shared__ float p2l[NCLS];
  __shared__ float e2l[128];
  __shared__ int   lblS[128];
  __shared__ float wredL[8], wredC[8];

  const int tid  = threadIdx.x;
  const int lane = tid & 31;
  const int wv   = tid >> 5;
  const int h    = lane >> 4;
  const int l16  = lane & 15;
  const int blockBase = blockIdx.x * 128;

#if __has_builtin(__builtin_amdgcn_tensor_load_to_lds) && __has_builtin(__builtin_amdgcn_s_wait_tensorcnt)
  // --- TDM staging: 2D tensor 512x64 of 2B elems, LDS padding 4 dwords every
  // 256 dwords (one row) -> reproduces the 520-element padded LDS stride. ---
  if (tid < 32) {   // one wave issues the DMA (EXEC ignored by TENSOR ops)
    typedef unsigned int u32x4 __attribute__((ext_vector_type(4)));
    typedef int i32x4 __attribute__((ext_vector_type(4)));
    typedef int i32x8 __attribute__((ext_vector_type(8)));
    i32x8 g1;
    g1[0] = (1 << 16)      // data_size = 2 bytes
          | (1 << 20)      // pad_enable
          | (7 << 22)      // pad_interval = 256 dwords
          | (3 << 25);     // pad_amount   = 4 dwords (16 B)
    g1[1] = 512 << 16;     // tensor_dim0[15:0] = 512
    g1[2] = 64 << 16;      // tensor_dim1[15:0] = 64
    g1[3] = 512 << 16;     // tile_dim0 = 512
    g1[4] = 64;            // tile_dim1 = 64
    g1[5] = 512;           // tensor_dim0_stride = 512
    g1[6] = 0; g1[7] = 0;
    i32x4 gz; gz[0] = gz[1] = gz[2] = gz[3] = 0;   // no dims 2..4
    const unsigned long long gaH = (unsigned long long)(uintptr_t)pHi;
    const unsigned long long gaL = (unsigned long long)(uintptr_t)pLo;
    u32x4 g0h, g0l;
    g0h[0] = 1u;                                        // count = 1 valid D#
    g0h[1] = (unsigned)(uintptr_t)&BhiLds[0];           // lds_addr
    g0h[2] = (unsigned)(gaH & 0xffffffffu);             // global_addr[31:0]
    g0h[3] = (unsigned)((gaH >> 32) & 0x01ffffffu) | 0x80000000u;  // ga[56:32] | type=2
    g0l[0] = 1u;
    g0l[1] = (unsigned)(uintptr_t)&BloLds[0];
    g0l[2] = (unsigned)(gaL & 0xffffffffu);
    g0l[3] = (unsigned)((gaL >> 32) & 0x01ffffffu) | 0x80000000u;
#if __clang_major__ >= 23
    i32x8 z8; z8[0]=z8[1]=z8[2]=z8[3]=z8[4]=z8[5]=z8[6]=z8[7]=0;
    __builtin_amdgcn_tensor_load_to_lds(g0h, g1, gz, gz, z8, 0);
    __builtin_amdgcn_tensor_load_to_lds(g0l, g1, gz, gz, z8, 0);
#else
    __builtin_amdgcn_tensor_load_to_lds(g0h, g1, gz, gz, 0);
    __builtin_amdgcn_tensor_load_to_lds(g0l, g1, gz, gz, 0);
#endif
    __builtin_amdgcn_s_wait_tensorcnt(0);
  }
#else
  // --- Fallback: manual padded staging (float4 = 8 bf16 per chunk). ---
  {
    const float4* hg = (const float4*)pHi;
    const float4* lg = (const float4*)pLo;
    for (int i = tid; i < NCLS * (DIM / 8); i += 256) {
      const int c = i >> 6, d8 = i & 63;
      *(float4*)&BhiLds[c * BSTRIDE + d8 * 8] = hg[i];
      *(float4*)&BloLds[c * BSTRIDE + d8 * 8] = lg[i];
    }
  }
#endif
  if (tid < NCLS) p2l[tid] = p2[tid];
  if (tid < 128) {
    e2l[tid]  = e2[blockBase + tid];
    lblS[tid] = labels[blockBase + tid];
  }
  __syncthreads();

  const int rowBase = blockBase + wv * 16;
  const float* Arow = E + (size_t)(rowBase + l16) * DIM + 16 * h;

  v8f acc[4] = {};   // 4 n-tiles cover all 64 classes
#pragma unroll 2
  for (int k = 0; k < DIM; k += 32) {
    __builtin_prefetch(Arow + k + 128, 0, 3);
    // 16 fp32 per lane: one contiguous 16-element K run of this lane's row.
    const float4 f0 = *(const float4*)(Arow + k);
    const float4 f1 = *(const float4*)(Arow + k + 4);
    const float4 f2 = *(const float4*)(Arow + k + 8);
    const float4 f3 = *(const float4*)(Arow + k + 12);
    const float xs[16] = {f0.x, f0.y, f0.z, f0.w, f1.x, f1.y, f1.z, f1.w,
                          f2.x, f2.y, f2.z, f2.w, f3.x, f3.y, f3.z, f3.w};
    v16bf ahi, alo;
#pragma unroll
    for (int e = 0; e < 16; ++e) {
      const __bf16 hb = f2bf(xs[e]);
      ahi[e] = hb;
      alo[e] = f2bf(xs[e] - bf2f(hb));
    }
    const int kb = k + 16 * h;
#pragma unroll
    for (int t = 0; t < 4; ++t) {
      const v16bf bhi = *(const v16bf*)&BhiLds[(t * 16 + l16) * BSTRIDE + kb];
      const v16bf blo = *(const v16bf*)&BloLds[(t * 16 + l16) * BSTRIDE + kb];
      acc[t] = __builtin_amdgcn_wmma_f32_16x16x32_bf16(
          false, ahi, false, bhi, (short)0, acc[t], false, false);
      acc[t] = __builtin_amdgcn_wmma_f32_16x16x32_bf16(
          false, alo, false, bhi, (short)0, acc[t], false, false);
      acc[t] = __builtin_amdgcn_wmma_f32_16x16x32_bf16(
          false, ahi, false, blo, (short)0, acc[t], false, false);
    }
  }

  // Epilogue: each 16-lane half owns one row's 64 logits per j-step.
  float myloss = 0.0f, mycorr = 0.0f;
#pragma unroll
  for (int j = 0; j < 8; ++j) {
    const int   rloc  = wv * 16 + j + 8 * h;
    const float e2row = e2l[rloc];
    float lg[4];
    float mx = -3.4e38f;
    int   am = 0;
#pragma unroll
    for (int t = 0; t < 4; ++t) {
      const int n = t * 16 + l16;
      float d2 = e2row + p2l[n] - 2.0f * acc[t][j];
      d2 = fmaxf(d2, 1e-12f);
      const float lv = -sqrtf(d2);
      lg[t] = lv;
      if (lv > mx) { mx = lv; am = n; }
    }
#pragma unroll
    for (int off = 1; off < 16; off <<= 1) {     // argmax, first-index tiebreak
      const float om = __shfl_xor(mx, off, 16);
      const int   oa = __shfl_xor(am, off, 16);
      if (om > mx || (om == mx && oa < am)) { mx = om; am = oa; }
    }
    float se = 0.0f;
#pragma unroll
    for (int t = 0; t < 4; ++t) se += expf(lg[t] - mx);
#pragma unroll
    for (int off = 1; off < 16; off <<= 1) se += __shfl_xor(se, off, 16);
    const float lse = mx + logf(se);

    const int lb = lblS[rloc];
    if (l16 == (lb & 15)) {                      // this lane holds column lb
      const int tt = lb >> 4;
      const float lgl = (tt == 0) ? lg[0] : (tt == 1) ? lg[1]
                      : (tt == 2) ? lg[2] : lg[3];
      myloss += lse - lgl;
    }
    if (l16 == 0) mycorr += (am == lb) ? 1.0f : 0.0f;
  }

#pragma unroll
  for (int off = 1; off < 32; off <<= 1) {
    myloss += __shfl_xor(myloss, off, 32);
    mycorr += __shfl_xor(mycorr, off, 32);
  }
  if (lane == 0) { wredL[wv] = myloss; wredC[wv] = mycorr; }
  __syncthreads();
  if (tid == 0) {                                // ordered -> deterministic
    float L = 0.0f, Cq = 0.0f;
#pragma unroll
    for (int w = 0; w < 8; ++w) { L += wredL[w]; Cq += wredC[w]; }
    partLoss[blockIdx.x] = L;
    partCorr[blockIdx.x] = Cq;
  }
}

// ---------------------------------------------------------------------------
// Kernel 5: sequential (deterministic) final reduction over 512 partials.
// ---------------------------------------------------------------------------
__global__ void final_kernel(const float* __restrict__ pl,
                             const float* __restrict__ pc,
                             float* __restrict__ out, int nb) {
  if (threadIdx.x == 0 && blockIdx.x == 0) {
    float L = 0.0f, Cq = 0.0f;
    for (int i = 0; i < nb; ++i) { L += pl[i]; Cq += pc[i]; }
    const float invN = 1.0f / (float)N_ROWS;
    out[0] = L * invN;   // mean NLL
    out[1] = Cq * invN;  // accuracy
  }
}

extern "C" void kernel_launch(void* const* d_in, const int* in_sizes, int n_in,
                              void* d_out, int out_size, void* d_ws, size_t ws_size,
                              hipStream_t stream) {
  const float* E      = (const float*)d_in[0];
  const int*   labels = (const int*)d_in[1];
  // d_in[2] = num_classes scalar on device; shapes fixed per reference (C=64).

  float* ws     = (float*)d_ws;
  float* proto  = ws;                        // 64*512 fp32
  float* counts = proto + NCLS * DIM;        // 64
  float* p2     = counts + NCLS;             // 64
  float* e2     = p2 + NCLS;                 // 65536
  float* pl     = e2 + N_ROWS;               // 512
  float* pc     = pl + (N_ROWS / 128);       // 512
  __hip_bfloat16* pHi = (__hip_bfloat16*)(pc + (N_ROWS / 128));  // 64*512 bf16 (16B-aligned)
  __hip_bfloat16* pLo = pHi + NCLS * DIM;                        // 64*512 bf16
  float* out    = (float*)d_out;

  proto_sum_kernel<<<dim3(NCLS, 4), 128, 0, stream>>>(E, labels, proto, counts);
  proto_finalize_kernel<<<NCLS, 128, 0, stream>>>(proto, counts, p2, pHi, pLo);
  e2_kernel<<<N_ROWS / 8, 256, 0, stream>>>(E, e2);
  cross_softmax_kernel<<<N_ROWS / 128, 256, 0, stream>>>(E, labels, pHi, pLo, p2, e2, pl, pc);
  final_kernel<<<1, 32, 0, stream>>>(pl, pc, out, N_ROWS / 128);
}
